// HawkesProcess_31756988186661
// MI455X (gfx1250) — compile-verified, matrix operand order checked by reading, not run on hardware
//
#include <hip/hip_runtime.h>
#include <hip/hip_bf16.h>

// Problem constants (match reference)
#define N_EV   1024
#define M_PAST 2048
#define T_GRID 64
#define G_GRID 512
#define D_COV  16
#define TWO_PI_F 6.2831853071795864f
#define EPS_F  1e-6f

// fp32 WMMA confirmed lowering to v_wmma_f32_16x16x4_f32 on gfx1250 (round 1).
// Keep the codegen-confirmed f16 fallback for toolchains lacking the builtin.
#if __has_builtin(__builtin_amdgcn_wmma_f32_16x16x4_f32)
#define USE_F32X4 1
#else
#define USE_F32X4 0
#endif

typedef __attribute__((ext_vector_type(2)))  float    v2f;
typedef __attribute__((ext_vector_type(8)))  float    v8f;
typedef __attribute__((ext_vector_type(8)))  _Float16 v8h;
typedef __attribute__((ext_vector_type(16))) _Float16 v16h;

// ---- workspace layout (carved on device; host never needs it) ----
// floats [0..127]   : base-intensity per-block partial sums
// floats [128..255] : excitation per-WMMA-wave partial sums (split-K partials)
// then S matrix [g*N+n] (G*N elems), then W matrix [t*N+n] (T*N elems)
__device__ __forceinline__ float* ws_partials(void* ws) { return (float*)ws; }
#if USE_F32X4
__device__ __forceinline__ float* ws_S(void* ws) { return (float*)ws + 256; }
__device__ __forceinline__ float* ws_W(void* ws) { return (float*)ws + 256 + (size_t)G_GRID * N_EV; }
#else
__device__ __forceinline__ _Float16* ws_S(void* ws) { return (_Float16*)((float*)ws + 256); }
__device__ __forceinline__ _Float16* ws_W(void* ws) { return ws_S(ws) + (size_t)G_GRID * N_EV; }
#endif

// ---------------------------------------------------------------------------
// Kernel 1: per-event log intensity.  One block per event, streams M=2048
// past events (24 MB total traffic -> HBM-bound, ~1 us at 23.3 TB/s).
// Both exponentials fused: alpha*w/(2*pi*s^2) * exp(-(r2/(2 s^2) + w*dt)).
// ---------------------------------------------------------------------------
__global__ void __launch_bounds__(256)
hawkes_event_kernel(const float* __restrict__ x, const float* __restrict__ t,
                    const float* __restrict__ past_x, const float* __restrict__ past_t,
                    const float* __restrict__ cov, const float* __restrict__ beta,
                    const float* __restrict__ alpha_p, const float* __restrict__ sigma_p,
                    const float* __restrict__ omega_p, float* __restrict__ out) {
    __shared__ float red[256];
    const int n = blockIdx.x;
    const float alpha = *alpha_p, sigma = *sigma_p, omega = *omega_p;
    const float inv2s2 = 1.0f / (2.0f * sigma * sigma);
    const float coef   = alpha * omega / (TWO_PI_F * sigma * sigma);
    const float xn0 = x[2 * n], xn1 = x[2 * n + 1];
    const float tn  = t[n];
    const float* __restrict__ px = past_x + (size_t)n * M_PAST * 2;
    const float* __restrict__ pt = past_t + (size_t)n * M_PAST;

    float s = 0.0f;
    for (int m = threadIdx.x; m < M_PAST; m += 256) {
        const float dt  = tn - pt[m];
        const v2f   pxy = *(const v2f*)(px + 2 * m);
        const float dx0 = xn0 - pxy[0];
        const float dx1 = xn1 - pxy[1];
        const float r2  = dx0 * dx0 + dx1 * dx1;
        const float dtc = dt > 0.0f ? dt : 0.0f;
        const float e   = __expf(-(r2 * inv2s2 + omega * dtc));
        s += (dt > 0.0f) ? e : 0.0f;
    }
    red[threadIdx.x] = s;
    __syncthreads();
    for (int st = 128; st > 0; st >>= 1) {
        if (threadIdx.x < st) red[threadIdx.x] += red[threadIdx.x + st];
        __syncthreads();
    }
    if (threadIdx.x == 0) {
        float mu = 0.0f;
#pragma unroll
        for (int d = 0; d < D_COV; ++d) mu += cov[(size_t)n * D_COV + d] * beta[d];
        mu = fmaxf(mu, EPS_F);
        out[n] = __logf(mu + coef * red[0] + EPS_F);
    }
}

// ---------------------------------------------------------------------------
// Kernel 2: spatial factor S[g*N+n] = exp(-r2/(2 s^2)) / (2 pi s^2)
// (depends only on (n,g) -> N*G = 512K exps instead of N*T*G = 33.5M)
// ---------------------------------------------------------------------------
__global__ void __launch_bounds__(256)
spatial_kernel(const float* __restrict__ x, const float* __restrict__ x_grid,
               const float* __restrict__ sigma_p, void* ws) {
    const int i = blockIdx.x * 256 + threadIdx.x;  // i = g*N + n
    const int n = i & (N_EV - 1);
    const int g = i >> 10;
    const float sigma = *sigma_p;
    const float inv2s2   = 1.0f / (2.0f * sigma * sigma);
    const float inv2pis2 = 1.0f / (TWO_PI_F * sigma * sigma);
    const float dx0 = x[2 * n]     - x_grid[2 * g];
    const float dx1 = x[2 * n + 1] - x_grid[2 * g + 1];
    const float r2  = dx0 * dx0 + dx1 * dx1;
    const float val = __expf(-r2 * inv2s2) * inv2pis2;
#if USE_F32X4
    ws_S(ws)[i] = val;
#else
    ws_S(ws)[i] = (_Float16)val;
#endif
}

// ---------------------------------------------------------------------------
// Kernel 3: temporal factor W[t*N+n] = alpha*omega*exp(-omega*dt) * [dt>0]
// (the mask lives entirely in this factor, so the S*W factorization is exact)
// ---------------------------------------------------------------------------
__global__ void __launch_bounds__(256)
temporal_kernel(const float* __restrict__ t, const float* __restrict__ t_grid,
                const float* __restrict__ alpha_p, const float* __restrict__ omega_p,
                void* ws) {
    const int i  = blockIdx.x * 256 + threadIdx.x;  // i = t_idx*N + n
    const int n  = i & (N_EV - 1);
    const int ti = i >> 10;
    const float alpha = *alpha_p, omega = *omega_p;
    const float dt  = t_grid[ti] - t[n];
    const float val = (dt > 0.0f) ? alpha * omega * __expf(-omega * dt) : 0.0f;
#if USE_F32X4
    ws_W(ws)[i] = val;
#else
    ws_W(ws)[i] = (_Float16)val;
#endif
}

// ---------------------------------------------------------------------------
// Kernel 4: baseline integral term: sum over T*G rows of max(z.beta, EPS).
// Deterministic per-block partials (no float atomics -> graph-stable).
// ---------------------------------------------------------------------------
__global__ void __launch_bounds__(256)
base_kernel(const float* __restrict__ z_flat, const float* __restrict__ beta, void* ws) {
    __shared__ float red[256];
    const int i = blockIdx.x * 256 + threadIdx.x;  // row < T*G = 32768
    const float* __restrict__ zr = z_flat + (size_t)i * D_COV;
    float dot = 0.0f;
#pragma unroll
    for (int d = 0; d < D_COV; ++d) dot += zr[d] * beta[d];
    red[threadIdx.x] = fmaxf(dot, EPS_F);
    __syncthreads();
    for (int st = 128; st > 0; st >>= 1) {
        if (threadIdx.x < st) red[threadIdx.x] += red[threadIdx.x + st];
        __syncthreads();
    }
    if (threadIdx.x == 0) ws_partials(ws)[blockIdx.x] = red[0];
}

// ---------------------------------------------------------------------------
// Kernel 5: WMMA GEMM, C[g,t] = sum_n S^T[g,n] * W[n,t]  (512 x 64 x 1024 f32).
//
// Split-K + A-reuse tiling: 128 waves = 32 g-tiles x 4 K-chunks.  Each wave
// owns a 16(g) x 64(t = all 4 t-tiles) strip over a 256-wide K chunk, keeping
// 4 f32 accumulators so each A fragment load feeds 4 WMMAs:
//   5 x global_load_b64 per 4 x v_wmma_f32_16x16x4_f32  (vs 16 before).
// Fragment layout per ISA 7.12.2 (16x4 f32 A): lanes 0-15 rows, VGPR0/1 hold
// K={0,1} (lanes<16) / K={2,3} (lanes>=16) -> contiguous 8-byte lane loads
// from S stored g-major / W stored t-major.  Only the total tile sum is
// needed (the integral sums everything), so the 32 accumulator VGPRs reduce
// through a wave shuffle into one deterministic partial per wave.
// __builtin_prefetch on the streaming S row exercises the CDNA5
// global_prefetch_b8 path (speculative RT, non-faulting, ISA 10.5).
// ---------------------------------------------------------------------------
__global__ void __launch_bounds__(128)
excite_wmma_kernel(void* ws) {
    const int lane = threadIdx.x & 31;
    const int wave = threadIdx.x >> 5;
    const int w    = blockIdx.x * 4 + wave;   // 0..127
    const int gt   = w >> 2;                  // 32 g-tiles
    const int kc   = w & 3;                   // 4 split-K chunks
    const int row  = lane & 15;
    const int hi   = lane >> 4;
    const int KC   = N_EV / 4;                // 256
    const int k0   = kc * KC;

    v8f c0 = {}, c1 = {}, c2 = {}, c3 = {};
#if USE_F32X4
    const float* __restrict__ Srow = ws_S(ws) + (size_t)(gt * 16 + row) * N_EV + k0 + 2 * hi;
    const float* __restrict__ Wb   = ws_W(ws) + (size_t)row * N_EV + k0 + 2 * hi;
#pragma unroll 2
    for (int k = 0; k < KC; k += 4) {
        __builtin_prefetch(Srow + k + 128, 0, 1);           // global_prefetch_b8
        const v2f a  = *(const v2f*)(Srow + k);
        const v2f b0 = *(const v2f*)(Wb + k);
        const v2f b1 = *(const v2f*)(Wb + 16 * N_EV + k);
        const v2f b2 = *(const v2f*)(Wb + 32 * N_EV + k);
        const v2f b3 = *(const v2f*)(Wb + 48 * N_EV + k);
        c0 = __builtin_amdgcn_wmma_f32_16x16x4_f32(false, a, false, b0, (short)0, c0, false, false);
        c1 = __builtin_amdgcn_wmma_f32_16x16x4_f32(false, a, false, b1, (short)0, c1, false, false);
        c2 = __builtin_amdgcn_wmma_f32_16x16x4_f32(false, a, false, b2, (short)0, c2, false, false);
        c3 = __builtin_amdgcn_wmma_f32_16x16x4_f32(false, a, false, b3, (short)0, c3, false, false);
    }
#else
    // f16 fallback: 16-bit A layout (ISA 7.12.2): two contiguous 8-half K-runs
    // per lane, K = 8*hi..8*hi+7 and 16+8*hi..16+8*hi+7.
    const _Float16* __restrict__ Srow = ws_S(ws) + (size_t)(gt * 16 + row) * N_EV + k0;
    const _Float16* __restrict__ Wb   = ws_W(ws) + (size_t)row * N_EV + k0;
    for (int k = 0; k < KC; k += 32) {
        const v8h a0 = *(const v8h*)(Srow + k + 8 * hi);
        const v8h a1 = *(const v8h*)(Srow + k + 16 + 8 * hi);
        v16h a;
#pragma unroll
        for (int i = 0; i < 8; ++i) { a[i] = a0[i]; a[8 + i] = a1[i]; }
#pragma unroll
        for (int tt = 0; tt < 4; ++tt) {
            const _Float16* __restrict__ Wr = Wb + (size_t)tt * 16 * N_EV;
            const v8h b0 = *(const v8h*)(Wr + k + 8 * hi);
            const v8h b1 = *(const v8h*)(Wr + k + 16 + 8 * hi);
            v16h b;
#pragma unroll
            for (int i = 0; i < 8; ++i) { b[i] = b0[i]; b[8 + i] = b1[i]; }
            v8f* cp = (tt == 0) ? &c0 : (tt == 1) ? &c1 : (tt == 2) ? &c2 : &c3;
            *cp = __builtin_amdgcn_wmma_f32_16x16x32_f16(false, a, false, b, (short)0, *cp,
                                                         false, false);
        }
    }
#endif
    float s = 0.0f;
#pragma unroll
    for (int i = 0; i < 8; ++i) s += c0[i] + c1[i] + c2[i] + c3[i];
#pragma unroll
    for (int off = 16; off > 0; off >>= 1) s += __shfl_xor(s, off, 32);
    if (lane == 0) ws_partials(ws)[128 + w] = s;
}

// ---------------------------------------------------------------------------
// Kernel 6: finalize -> out[N] = (base_sum + excite_sum) * (1/G) * dt_step
// ---------------------------------------------------------------------------
__global__ void finalize_kernel(const float* __restrict__ t_grid, void* ws,
                                float* __restrict__ out) {
    float s = 0.0f;
    const float* p = ws_partials(ws);
    for (int i = 0; i < 256; ++i) s += p[i];
    const float dt_step = t_grid[1] - t_grid[0];
    out[N_EV] = s * (1.0f / (float)G_GRID) * dt_step;
}

extern "C" void kernel_launch(void* const* d_in, const int* in_sizes, int n_in,
                              void* d_out, int out_size, void* d_ws, size_t ws_size,
                              hipStream_t stream) {
    const float* x       = (const float*)d_in[0];
    const float* t       = (const float*)d_in[1];
    const float* past_x  = (const float*)d_in[2];
    const float* past_t  = (const float*)d_in[3];
    const float* cov     = (const float*)d_in[4];
    const float* z_grid  = (const float*)d_in[5];
    const float* x_grid  = (const float*)d_in[6];
    const float* t_grid  = (const float*)d_in[7];
    const float* beta    = (const float*)d_in[8];
    const float* alpha   = (const float*)d_in[9];
    const float* sigma   = (const float*)d_in[10];
    const float* omega   = (const float*)d_in[11];
    float* out = (float*)d_out;

    (void)in_sizes; (void)n_in; (void)out_size; (void)ws_size;

    // log intensities at events
    hawkes_event_kernel<<<N_EV, 256, 0, stream>>>(x, t, past_x, past_t, cov, beta,
                                                  alpha, sigma, omega, out);
    // factor matrices for the integral GEMM
    spatial_kernel<<<(G_GRID * N_EV) / 256, 256, 0, stream>>>(x, x_grid, sigma, d_ws);
    temporal_kernel<<<(T_GRID * N_EV) / 256, 256, 0, stream>>>(t, t_grid, alpha, omega, d_ws);
    // baseline term partials
    base_kernel<<<(T_GRID * G_GRID) / 256, 256, 0, stream>>>(z_grid, beta, d_ws);
    // WMMA excitation GEMM partials (split-K, A-reuse)
    excite_wmma_kernel<<<32, 128, 0, stream>>>(d_ws);
    // combine
    finalize_kernel<<<1, 1, 0, stream>>>(t_grid, d_ws, out);
}
